// LSTMModel_49675591746257
// MI455X (gfx1250) — matrix operand order repaired and checked by reading, not compiled, not run on hardware
//
#include <hip/hip_runtime.h>
#include <stdint.h>

// Problem dims (fixed by the reference)
#define BATCH 64
#define SEQ   512
#define IN    512
#define HID   1024
#define OUTN  512
#define GATES 4096   // 4*HID

typedef __bf16 bf16;
typedef __attribute__((ext_vector_type(16))) __bf16 v16bf;
typedef __attribute__((ext_vector_type(8)))  __bf16 v8bf;
typedef __attribute__((ext_vector_type(8)))  float  v8f;

// LDS staging geometry for the step kernel.
// Each K-chunk buffer: A = 64 rows x 32 halves, B = 128 rows x 32 halves.
// Rows padded 64B -> 80B (20 dwords): 16 lane row-starts tile all 64 LDS banks.
#define ROWPITCH 80
#define A_ROWS   64
#define B_ROWS   128
#define B_BASE   (A_ROWS * ROWPITCH)                 // 5120
#define BUF_SZ   (B_BASE + B_ROWS * ROWPITCH)        // 15360
#define NBUF     3                                   // 2 chunks in flight
#define NCHUNK1  (IN / 32)                           // 16
#define NCHUNK2  (HID / 32)                          // 32

// Rotating LDS buffer offset (avoids integer modulo in the hot loop)
#define ADV(o) do { (o) += BUF_SZ; if ((o) == NBUF * BUF_SZ) (o) = 0; } while (0)

// ---------------------------------------------------------------------------
// WMMA helpers (CDNA5 wave32, V_WMMA_F32_16X16X32_BF16)
// A (16x32, MxK): lane m holds row m&15, 16 contiguous K halves
//                 starting at k0 + 16*(lane>>4).
// B (32x16, KxN): lane n holds column n (= W row j0+n), 16 contiguous K halves
//                 starting at k0 + 16*(lane>>4).
// C/D (16x16 f32): vgpr r, lanes 0-15 -> M=r ; lanes 16-31 -> M=8+r ; N=lane&15.
// ---------------------------------------------------------------------------
__device__ __forceinline__ v16bf cat8(v8bf lo, v8bf hi) {
  return __builtin_shufflevector(lo, hi, 0, 1, 2, 3, 4, 5, 6, 7,
                                         8, 9, 10, 11, 12, 13, 14, 15);
}

__device__ __forceinline__ v16bf load_a_global(const bf16* __restrict__ rowPtr,
                                               int k0, int laneHalf) {
  const v8bf lo = *(const v8bf*)(rowPtr + k0 + 16 * laneHalf);
  const v8bf hi = *(const v8bf*)(rowPtr + k0 + 8 + 16 * laneHalf);
  return cat8(lo, hi);
}

__device__ __forceinline__ v16bf load_b_global(const bf16* __restrict__ colPtr,
                                               int k0, int laneHalf) {
  return *(const v16bf*)(colPtr + k0 + 16 * laneHalf);
}

// LDS fragment reads (chunk-local K in [0,32))
__device__ __forceinline__ v16bf lds_read_a(const char* buf, int row,
                                            int laneHalf) {
  const char* p = buf + row * ROWPITCH + laneHalf * 32;
  return cat8(*(const v8bf*)p, *(const v8bf*)(p + 16));
}
__device__ __forceinline__ v16bf lds_read_b(const char* buf, int row,
                                            int laneHalf) {
  const char* p = buf + B_BASE + row * ROWPITCH + laneHalf * 32;
  return cat8(*(const v8bf*)p, *(const v8bf*)(p + 16));
}

__device__ __forceinline__ v8f wmma_bf16(v16bf a, v16bf b, v8f c) {
  return __builtin_amdgcn_wmma_f32_16x16x32_bf16(
      /*neg_a=*/false, a, /*neg_b=*/false, b,
      /*c_mod=*/(short)0, c, /*reuse_a=*/false, /*reuse_b=*/false);
}

__device__ __forceinline__ float sigmoidf_fast(float x) {
  return 1.0f / (1.0f + __expf(-x));
}
__device__ __forceinline__ float tanhf_fast(float x) {
  const float e = __expf(-2.0f * x);
  return (1.0f - e) / (1.0f + e);
}

// ---------------------------------------------------------------------------
// CDNA5 async global -> LDS copy (16 bytes per lane), tracked by ASYNCcnt.
// Async loads complete in order per wave, so waiting ASYNCcnt<=3 guarantees
// everything except the most recent stage3() (exactly 3 loads) has landed.
// ---------------------------------------------------------------------------
__device__ __forceinline__ void async_copy16(uint32_t lds_off, const void* g) {
  asm volatile("global_load_async_to_lds_b128 %0, %1, off"
               :: "v"(lds_off), "v"(g)
               : "memory");
}
__device__ __forceinline__ void wait_async0() {
  asm volatile("s_wait_asynccnt 0x0" ::: "memory");
}
__device__ __forceinline__ void wait_async3() {
  asm volatile("s_wait_asynccnt 0x3" ::: "memory");
}

// Stage this thread's 3 x 16B segments of one 32-deep K-chunk:
//  A: 64 batch rows x 64B  (one segment per thread)
//  B: 128 W rows (4 gates x 32 cols) x 64B (two adjacent segments per thread)
__device__ __forceinline__ void stage3(uint32_t bufBase, uint32_t ldsAoff,
                                       uint32_t ldsBoff,
                                       const bf16* __restrict__ ap,
                                       const bf16* __restrict__ bp) {
  async_copy16(bufBase + ldsAoff, ap);
  async_copy16(bufBase + ldsBoff, bp);
  async_copy16(bufBase + ldsBoff + 16, bp + 8);
}

// Loads-then-MMAs: all 10 ds_loads must issue before any WMMA (enforced with a
// scheduling barrier); DS returns in order so the compiler can use partial
// dscnt waits, and the 4 WMMAs (disjoint accumulators) run back-to-back.
__device__ __forceinline__ void compute_chunk(const char* buf, int rowA,
                                              int bcol, int laneHalf,
                                              v8f (&acc)[4]) {
  const v16bf a  = lds_read_a(buf, rowA, laneHalf);
  const v16bf b0 = lds_read_b(buf, 0 * 32 + bcol, laneHalf);
  const v16bf b1 = lds_read_b(buf, 1 * 32 + bcol, laneHalf);
  const v16bf b2 = lds_read_b(buf, 2 * 32 + bcol, laneHalf);
  const v16bf b3 = lds_read_b(buf, 3 * 32 + bcol, laneHalf);
  __builtin_amdgcn_sched_barrier(0);  // keep all ds_loads above the WMMAs
  acc[0] = wmma_bf16(a, b0, acc[0]);
  acc[1] = wmma_bf16(a, b1, acc[1]);
  acc[2] = wmma_bf16(a, b2, acc[2]);
  acc[3] = wmma_bf16(a, b3, acc[3]);
}

// ---------------------------------------------------------------------------
// f32 -> bf16 cast (grid-stride)
// ---------------------------------------------------------------------------
__global__ void f32_to_bf16_kernel(const float* __restrict__ src,
                                   bf16* __restrict__ dst, int n) {
  int i = blockIdx.x * blockDim.x + threadIdx.x;
  const int stride = gridDim.x * blockDim.x;
  for (; i < n; i += stride) dst[i] = (bf16)src[i];
}

// ---------------------------------------------------------------------------
// One LSTM timestep: g = x_t @ W_ih^T + h_{s-1} @ W_hh^T + b_ih + b_hh.
// Grid: 32 blocks x 256 threads. Block b owns hidden cols [32b, 32b+32) for
// all 4 gates and all 64 batch rows. Operands are staged into a 3-buffer LDS
// pipeline with async loads; 2 chunks stay in flight over the WMMA compute.
// Steady-state iterations use an unconditional s_wait_asynccnt 3; the last
// chunk is peeled into an epilogue with s_wait_asynccnt 0.
// Race-freedom: iteration g computes buffer cOff; stage(g+2) rewrites the
// buffer read in iteration g-1 only after the barrier at the top of iteration
// g, which all readers of that buffer have passed.
// ---------------------------------------------------------------------------
__global__ __launch_bounds__(256) void lstm_step_kernel(
    const bf16*  __restrict__ xb,    // [B][S][I]  bf16
    const bf16*  __restrict__ Wih,   // [4H][I]    bf16
    const bf16*  __restrict__ Whh,   // [4H][H]    bf16
    const float* __restrict__ b_ih,  // [4H]
    const float* __restrict__ b_hh,  // [4H]
    bf16*        __restrict__ hs,    // [S][B][H]  bf16 history
    float*       __restrict__ Cbuf,  // [B][H]     fp32 cell state
    float*       __restrict__ hf,    // [B][H]     fp32 latest h (for h_f)
    int s) {
  __shared__ __align__(16) char smem[NBUF * BUF_SZ];

  const int tid      = threadIdx.x;
  const int lane     = tid & 31;
  const int wave     = tid >> 5;
  const int laneHalf = lane >> 4;
  const int lane15   = lane & 15;
  const int tM       = wave & 3;       // batch tile 0..3
  const int tNl      = wave >> 2;      // local hidden tile 0..1
  const int jblk     = blockIdx.x * 32;
  const int j0       = jblk + tNl * 16;
  const int brow0    = tM * 16;
  const int rowA16   = brow0 + lane15;     // A fragment row for this lane
  const int bcol     = tNl * 16 + lane15;  // local B column for this lane

  // Per-thread staging constants
  const int rowA = tid >> 2, segA = tid & 3;       // A: 64 rows x 4 segs
  const int rbB  = tid >> 1, segB = (tid & 1) * 2; // B: 128 rows x 2x2 segs
  const int colB = (rbB >> 5) * HID + jblk + (rbB & 31);
  const uint32_t ldsAoff = (uint32_t)(rowA * ROWPITCH + segA * 16);
  const uint32_t ldsBoff = (uint32_t)(B_BASE + rbB * ROWPITCH + segB * 16);

  // LDS byte offset of smem (truncate the generic address: low 32 bits of the
  // LDS-aperture flat address are the wave-relative LDS offset).
  const uint32_t smemBase = (uint32_t)(uintptr_t)(void*)smem;

  // Per-thread global base pointers for each phase (advance 32 halves/chunk)
  const bf16* aP1 = xb + ((size_t)rowA * SEQ + s) * IN + segA * 8;
  const bf16* bP1 = Wih + (size_t)colB * IN + segB * 8;
  const int sPrev = (s > 0) ? (s - 1) : 0;  // guarded; only used when s>0
  const bf16* aP2 = hs + ((size_t)sPrev * BATCH + rowA) * HID + segA * 8;
  const bf16* bP2 = Whh + (size_t)colB * HID + segB * 8;

  v8f acc[4];  // gate order: i, f, c, o  (rows of W: gate*H + j)
#pragma unroll
  for (int g = 0; g < 4; ++g) {
    const int col = g * HID + j0 + lane15;
    const float bias = b_ih[col] + b_hh[col];
#pragma unroll
    for (int r = 0; r < 8; ++r) acc[g][r] = bias;
  }

  const int nk = (s > 0) ? (NCHUNK1 + NCHUNK2) : NCHUNK1;

  // Prologue: chunks 0 and 1 (both phase 1; NCHUNK1 >= 2)
  stage3(smemBase + 0 * BUF_SZ, ldsAoff, ldsBoff, aP1, bP1);
  stage3(smemBase + 1 * BUF_SZ, ldsAoff, ldsBoff, aP1 + 32, bP1 + 32);

  uint32_t cOff = 0;                 // compute-buffer byte offset (chunk g)
  uint32_t sOff = 2 * BUF_SZ;        // stage-buffer byte offset  (chunk g+2)

  // Phase 1 steady: all phase-1 chunks except, at s==0, the final one
  const int p1Steady = (s > 0) ? NCHUNK1 : (NCHUNK1 - 1);
  for (int kc = 0; kc < p1Steady; ++kc) {
    wait_async3();
    __syncthreads();
    const int nxt = kc + 2;
    if (nxt < NCHUNK1) {
      stage3(smemBase + sOff, ldsAoff, ldsBoff, aP1 + nxt * 32, bP1 + nxt * 32);
    } else if (s > 0 && nxt < nk) {
      const int k2 = nxt - NCHUNK1;
      stage3(smemBase + sOff, ldsAoff, ldsBoff, aP2 + k2 * 32, bP2 + k2 * 32);
    }
    compute_chunk(smem + cOff, rowA16, bcol, laneHalf, acc);
    ADV(cOff);
    ADV(sOff);
  }
  // Phase 2 steady (s>0): chunks 16..46 (last chunk peeled)
  if (s > 0) {
    for (int kc = 0; kc < NCHUNK2 - 1; ++kc) {
      wait_async3();
      __syncthreads();
      const int nxt = kc + 2;
      if (nxt < NCHUNK2)
        stage3(smemBase + sOff, ldsAoff, ldsBoff, aP2 + nxt * 32,
               bP2 + nxt * 32);
      compute_chunk(smem + cOff, rowA16, bcol, laneHalf, acc);
      ADV(cOff);
      ADV(sOff);
    }
  }
  // Epilogue: final chunk (nk-1); drain ASYNCcnt fully
  wait_async0();
  __syncthreads();
  compute_chunk(smem + cOff, rowA16, bcol, laneHalf, acc);

  // Gate nonlinearities + state update (each (b, j) owned by one lane/reg)
  const int j = j0 + lane15;
#pragma unroll
  for (int r = 0; r < 8; ++r) {
    const int brow = brow0 + r + 8 * laneHalf;
    const size_t idx = (size_t)brow * HID + j;
    const float ig = sigmoidf_fast(acc[0][r]);
    const float fg = sigmoidf_fast(acc[1][r]);
    const float cg = tanhf_fast(acc[2][r]);
    const float og = sigmoidf_fast(acc[3][r]);
    const float Cp = (s == 0) ? 0.0f : Cbuf[idx];
    const float Cn = fg * Cp + ig * cg;
    const float hn = og * tanhf_fast(Cn);
    Cbuf[idx] = Cn;
    hf[idx]   = hn;
    hs[((size_t)s * BATCH + brow) * HID + j] = (bf16)hn;
  }
}

// ---------------------------------------------------------------------------
// Output projection: out[b][s][o] = hs[s][b][:] . W_out[o][:] + b_out[o]
// Each wave: one 16-row M tile (rows = s*64+b, all same s) x 64-col N strip.
// One-shot GEMM, HBM traffic trivial vs 23.3 TB/s -> direct global loads.
// ---------------------------------------------------------------------------
__global__ __launch_bounds__(256) void out_gemm_kernel(
    const bf16*  __restrict__ hs,    // [S][B][H]
    const bf16*  __restrict__ Wout,  // [O][H]
    const float* __restrict__ bout,  // [O]
    float*       __restrict__ out) { // [B][S][O]
  const int lane     = threadIdx.x & 31;
  const int wave     = threadIdx.x >> 5;
  const int waveId   = blockIdx.x * 8 + wave;
  const int strip    = waveId & 7;   // N strip of 64 (O=512 -> 8 strips)
  const int mt       = waveId >> 3;  // 0..2047 M tiles over S*B=32768
  const int laneHalf = lane >> 4;
  const int lane15   = lane & 15;
  const int m0       = mt * 16;
  const int sIdx     = m0 >> 6;      // m = s*64 + b ; 16 | 64 so tile has one s
  const int b0       = m0 & 63;
  const int n0       = strip * 64;

  v8f acc[4];
#pragma unroll
  for (int t = 0; t < 4; ++t) {
    const float bias = bout[n0 + t * 16 + lane15];
#pragma unroll
    for (int r = 0; r < 8; ++r) acc[t][r] = bias;
  }

  const bf16* arow = hs + ((size_t)sIdx * BATCH + b0 + lane15) * HID;
  const bf16* bcol0 = Wout + (size_t)(n0 + 0 * 16 + lane15) * HID;
  const bf16* bcol1 = Wout + (size_t)(n0 + 1 * 16 + lane15) * HID;
  const bf16* bcol2 = Wout + (size_t)(n0 + 2 * 16 + lane15) * HID;
  const bf16* bcol3 = Wout + (size_t)(n0 + 3 * 16 + lane15) * HID;
#pragma unroll 2
  for (int k = 0; k < HID; k += 32) {
    const v16bf a  = load_a_global(arow, k, laneHalf);
    const v16bf t0 = load_b_global(bcol0, k, laneHalf);
    const v16bf t1 = load_b_global(bcol1, k, laneHalf);
    const v16bf t2 = load_b_global(bcol2, k, laneHalf);
    const v16bf t3 = load_b_global(bcol3, k, laneHalf);
    acc[0] = wmma_bf16(a, t0, acc[0]);
    acc[1] = wmma_bf16(a, t1, acc[1]);
    acc[2] = wmma_bf16(a, t2, acc[2]);
    acc[3] = wmma_bf16(a, t3, acc[3]);
  }

#pragma unroll
  for (int t = 0; t < 4; ++t) {
    const int o = n0 + t * 16 + lane15;
#pragma unroll
    for (int r = 0; r < 8; ++r) {
      const int brow = b0 + r + 8 * laneHalf;
      out[((size_t)brow * SEQ + sIdx) * OUTN + o] = acc[t][r];
    }
  }
}

// ---------------------------------------------------------------------------
// Copy final h / C (kept in fp32 side buffers) into the output tail.
// ---------------------------------------------------------------------------
__global__ void copy_tail_kernel(const float* __restrict__ hf,
                                 const float* __restrict__ Cbuf,
                                 float* __restrict__ out) {
  const int i = blockIdx.x * blockDim.x + threadIdx.x;
  if (i < BATCH * HID) {
    const size_t base = (size_t)BATCH * SEQ * OUTN;
    out[base + i] = hf[i];
    out[base + (size_t)BATCH * HID + i] = Cbuf[i];
  }
}

// ---------------------------------------------------------------------------
// Host entry
// ---------------------------------------------------------------------------
extern "C" void kernel_launch(void* const* d_in, const int* in_sizes, int n_in,
                              void* d_out, int out_size, void* d_ws,
                              size_t ws_size, hipStream_t stream) {
  const float* x     = (const float*)d_in[0];
  const float* W_ih  = (const float*)d_in[1];
  const float* b_ih  = (const float*)d_in[2];
  const float* W_hh  = (const float*)d_in[3];
  const float* b_hh  = (const float*)d_in[4];
  const float* W_out = (const float*)d_in[5];
  const float* b_out = (const float*)d_in[6];
  float* out = (float*)d_out;

  // Workspace layout (bytes); all chunks 32B-aligned by construction.
  char* ws = (char*)d_ws;
  size_t off = 0;
  bf16* xb    = (bf16*)(ws + off); off += (size_t)BATCH * SEQ * IN * 2;   // 32 MB
  bf16* Wihb  = (bf16*)(ws + off); off += (size_t)GATES * IN * 2;        //  4 MB
  bf16* Whhb  = (bf16*)(ws + off); off += (size_t)GATES * HID * 2;       //  8 MB
  bf16* Woutb = (bf16*)(ws + off); off += (size_t)OUTN * HID * 2;        //  1 MB
  bf16* hsb   = (bf16*)(ws + off); off += (size_t)SEQ * BATCH * HID * 2; // 64 MB
  float* Cbuf = (float*)(ws + off); off += (size_t)BATCH * HID * 4;
  float* hf   = (float*)(ws + off); off += (size_t)BATCH * HID * 4;
  (void)ws_size; (void)in_sizes; (void)n_in; (void)out_size;

  // 1) Cast inputs/weights to bf16
  f32_to_bf16_kernel<<<512, 256, 0, stream>>>(x, xb, BATCH * SEQ * IN);
  f32_to_bf16_kernel<<<256, 256, 0, stream>>>(W_ih, Wihb, GATES * IN);
  f32_to_bf16_kernel<<<256, 256, 0, stream>>>(W_hh, Whhb, GATES * HID);
  f32_to_bf16_kernel<<<128, 256, 0, stream>>>(W_out, Woutb, OUTN * HID);

  // 2) Sequential scan: one launch per timestep (stream order carries the
  //    h/C dependency; W_ih+W_hh = 12 MB bf16 stays resident in 192 MB L2,
  //    async-LDS 3-buffer pipeline keeps 2 chunks in flight per wave)
  for (int s = 0; s < SEQ; ++s) {
    lstm_step_kernel<<<32, 256, 0, stream>>>(xb, Wihb, Whhb, b_ih, b_hh,
                                             hsb, Cbuf, hf, s);
  }

  // 3) Output projection over all timesteps
  out_gemm_kernel<<<2048, 256, 0, stream>>>(hsb, Woutb, b_out, out);

  // 4) Final h / C into output tail
  copy_tail_kernel<<<(BATCH * HID + 255) / 256, 256, 0, stream>>>(hf, Cbuf, out);
}
